// EncoderLayer_58884001628592
// MI455X (gfx1250) — matrix operand order repaired and checked by reading, not compiled, not run on hardware
//
#include <hip/hip_runtime.h>

// CDNA5 / gfx1250, wave32. Matrix math via v_wmma_f32_16x16x32_f16.
// Tile staging via TDM (tensor_load_to_lds) with LDS row padding done by the
// DMA engine (pad fields in D#), double-buffered, tracked with TENSORcnt.
// Weights pre-transposed so B fragments are K-contiguous; mask packed to bits.

typedef __attribute__((ext_vector_type(16))) _Float16 v16h;
typedef __attribute__((ext_vector_type(8)))  float    v8f;
typedef __attribute__((ext_vector_type(4)))  unsigned int u32x4;
typedef __attribute__((ext_vector_type(8)))  int      i32x8;
typedef __attribute__((ext_vector_type(4)))  int      i32x4;

#define BATCH    2
#define S_LEN    4096
#define D_MODEL  512
#define N_HEADS  8
#define D_HEAD   64
#define D_FF     2048
#define ROWS     (BATCH * S_LEN)   // 8192
#define SW       (S_LEN / 32)      // mask words per row

union Frag16 { v16h v; _Float16 h[16]; unsigned u[8]; u32x4 q[2]; };

__device__ __forceinline__ float grp16_max(float v) {
  v = fmaxf(v, __shfl_xor(v, 1, 32));
  v = fmaxf(v, __shfl_xor(v, 2, 32));
  v = fmaxf(v, __shfl_xor(v, 4, 32));
  v = fmaxf(v, __shfl_xor(v, 8, 32));
  return v;
}
__device__ __forceinline__ float grp16_sum(float v) {
  v += __shfl_xor(v, 1, 32);
  v += __shfl_xor(v, 2, 32);
  v += __shfl_xor(v, 4, 32);
  v += __shfl_xor(v, 8, 32);
  return v;
}

// ------------------------------------------------------------ TDM helper
// 2D f16 tile load: tile_y rows x tile_x elems, global row stride = stride
// elems, into LDS at lds_off with pad_amount=4 dwords inserted every
// (1<<pad_code) 8-byte units (pad_code 3 -> every 16 dwords, 4 -> every 32).
__device__ __forceinline__ void tdm_load_2d(unsigned lds_off, unsigned long long gaddr,
                                            unsigned tile_x, unsigned tile_y,
                                            unsigned long long stride_elems,
                                            unsigned pad_code) {
  u32x4 g0;
  g0.x = 1u;                                              // count=1, user D#
  g0.y = lds_off;                                         // lds_addr [63:32]
  g0.z = (unsigned)(gaddr & 0xFFFFFFFFull);               // global_addr lo
  g0.w = (unsigned)((gaddr >> 32) & 0x1FFFFFFull) | (2u << 30);  // hi + type=2
  i32x8 g1;
  g1[0] = (int)((1u << 16) | (1u << 20) | (pad_code << 22) | (3u << 25));
  g1[1] = (int)((tile_x & 0xFFFFu) << 16);                // tensor_dim0 lo16
  g1[2] = (int)(((tile_x >> 16) & 0xFFFFu) | ((tile_y & 0xFFFFu) << 16));
  g1[3] = (int)(((tile_y >> 16) & 0xFFFFu) | (tile_x << 16));   // tile_dim0
  g1[4] = (int)(tile_y & 0xFFFFu);                        // tile_dim1 (dim2=0)
  g1[5] = (int)(stride_elems & 0xFFFFFFFFull);            // dim0 stride lo32
  g1[6] = (int)((stride_elems >> 32) & 0xFFFFull);        // stride hi16
  g1[7] = 0;
  i32x4 gz = {0, 0, 0, 0};
#if __clang_major__ >= 23
  i32x8 gz8 = {0, 0, 0, 0, 0, 0, 0, 0};
  __builtin_amdgcn_tensor_load_to_lds(g0, g1, gz, gz, gz8, 0);
#else
  __builtin_amdgcn_tensor_load_to_lds(g0, g1, gz, gz, 0);
#endif
}

// ------------------------------------------------------------ weight prep
__global__ void convT_kernel(const float* __restrict__ in, _Float16* __restrict__ out,
                             int Kd, int Nd) {
  int idx = blockIdx.x * 256 + threadIdx.x;
  if (idx < Kd * Nd) {
    int k = idx / Nd, n = idx % Nd;
    out[(size_t)n * Kd + k] = (_Float16)in[idx];
  }
}

__global__ __launch_bounds__(256) void mask_pack_kernel(const int* __restrict__ mask,
                                                        unsigned* __restrict__ mb) {
  int lane = threadIdx.x & 31;
  int word = blockIdx.x * 8 + (threadIdx.x >> 5);
  int pred = mask[(long)word * 32 + lane] != 0;
  unsigned long long bb = __ballot(pred);
  if (lane == 0) mb[word] = (unsigned)bb;
}

// ------------------------------------------------------------ layernorm
__global__ __launch_bounds__(256) void ln_kernel(const float* __restrict__ x,
                                                 const float* __restrict__ a,
                                                 const float* __restrict__ b,
                                                 _Float16* __restrict__ out) {
  int wid = threadIdx.x >> 5, lane = threadIdx.x & 31;
  int row = blockIdx.x * 8 + wid;
  const float* xr = x + (long)row * D_MODEL;
  float v[16];
  float s = 0.0f;
#pragma unroll
  for (int i = 0; i < 16; ++i) { v[i] = xr[i * 32 + lane]; s += v[i]; }
#pragma unroll
  for (int o = 1; o < 32; o <<= 1) s += __shfl_xor(s, o, 32);
  float mean = s * (1.0f / 512.0f);
  float q = 0.0f;
#pragma unroll
  for (int i = 0; i < 16; ++i) { float d = v[i] - mean; q += d * d; }
#pragma unroll
  for (int o = 1; o < 32; o <<= 1) q += __shfl_xor(q, o, 32);
  float inv = 1.0f / (sqrtf(q * (1.0f / 511.0f)) + 1e-6f);
  _Float16* orow = out + (long)row * D_MODEL;
#pragma unroll
  for (int i = 0; i < 16; ++i) {
    int c = i * 32 + lane;
    orow[c] = (_Float16)(a[c] * (v[i] - mean) * inv + b[c]);
  }
}

// ------------------------------------------------------------ WMMA GEMM
// C[M,N] = A[M,K](f16,row) * BT[N,K](f16)  (+bias)(relu)(+resid)
// tile 128x64, BK=32; staging via TDM, double-buffered.
template <bool BIAS, bool RELU, bool RES, bool OUTF, bool OUTH, bool OUTHT>
__global__ __launch_bounds__(256) void gemm_wmma_kernel(
    const _Float16* __restrict__ A, const _Float16* __restrict__ BT,
    const float* __restrict__ bias, const float* __restrict__ resid,
    float* __restrict__ outF, _Float16* __restrict__ outH,
    _Float16* __restrict__ outHT, int M, int N, int K) {
  __shared__ u32x4 As4[2][128 * 5];   // row = 32 halfs + 16B pad (TDM-written)
  __shared__ u32x4 Bs4[2][64 * 5];

  int tid = threadIdx.x;
  int wid = tid >> 5, lane = tid & 31, lm = lane & 15, grp = lane >> 4;
  int wm = wid >> 1, wn = wid & 1;
  int m0 = blockIdx.y * 128, n0 = blockIdx.x * 64;
  bool leader = (wid == 0);

  unsigned aoff[2] = { (unsigned)(size_t)(void*)&As4[0][0],
                       (unsigned)(size_t)(void*)&As4[1][0] };
  unsigned boff[2] = { (unsigned)(size_t)(void*)&Bs4[0][0],
                       (unsigned)(size_t)(void*)&Bs4[1][0] };
  unsigned long long Abase = (unsigned long long)(A + (size_t)m0 * K);
  unsigned long long Bbase = (unsigned long long)(BT + (size_t)n0 * K);

  v8f acc[2][2] = {};

  if (leader) {
    tdm_load_2d(aoff[0], Abase, 32, 128, (unsigned)K, 3);
    tdm_load_2d(boff[0], Bbase, 32, 64, (unsigned)K, 3);
    __builtin_amdgcn_s_wait_tensorcnt(0);
  }
  __syncthreads();

  int cur = 0;
  for (int k0 = 0; k0 < K; k0 += 32) {
    bool more = (k0 + 32) < K;
    if (more && leader) {
      tdm_load_2d(aoff[1 - cur], Abase + (size_t)(k0 + 32) * 2, 32, 128, (unsigned)K, 3);
      tdm_load_2d(boff[1 - cur], Bbase + (size_t)(k0 + 32) * 2, 32, 64, (unsigned)K, 3);
    }
    const u32x4* Ab = As4[cur];
    const u32x4* Bb = Bs4[cur];
    Frag16 af[2], bf[2];
#pragma unroll
    for (int mi = 0; mi < 2; ++mi) {
      int r = wm * 32 + mi * 16 + lm;
      af[mi].q[0] = Ab[r * 5 + grp * 2];
      af[mi].q[1] = Ab[r * 5 + grp * 2 + 1];
    }
#pragma unroll
    for (int ni = 0; ni < 2; ++ni) {
      int c = wn * 32 + ni * 16 + lm;
      bf[ni].q[0] = Bb[c * 5 + grp * 2];
      bf[ni].q[1] = Bb[c * 5 + grp * 2 + 1];
    }
#pragma unroll
    for (int mi = 0; mi < 2; ++mi)
#pragma unroll
      for (int ni = 0; ni < 2; ++ni)
        acc[mi][ni] = __builtin_amdgcn_wmma_f32_16x16x32_f16(
            false, af[mi].v, false, bf[ni].v, (short)0, acc[mi][ni], false, false);
    if (more) {
      if (leader) __builtin_amdgcn_s_wait_tensorcnt(0);
      __syncthreads();
      cur ^= 1;
    }
  }

#pragma unroll
  for (int mi = 0; mi < 2; ++mi) {
#pragma unroll
    for (int ni = 0; ni < 2; ++ni) {
      int rbase = m0 + wm * 32 + mi * 16 + grp * 8;
      int c = n0 + wn * 32 + ni * 16 + lm;
      float bv = BIAS ? bias[c] : 0.0f;
      float vals[8];
#pragma unroll
      for (int j = 0; j < 8; ++j) {
        float val = acc[mi][ni][j] + bv;
        if (RELU) val = fmaxf(val, 0.0f);
        if (RES) val += resid[(size_t)(rbase + j) * N + c];
        vals[j] = val;
      }
      if (OUTF) {
#pragma unroll
        for (int j = 0; j < 8; ++j) outF[(size_t)(rbase + j) * N + c] = vals[j];
      }
      if (OUTH) {
#pragma unroll
        for (int j = 0; j < 8; ++j) outH[(size_t)(rbase + j) * N + c] = (_Float16)vals[j];
      }
      if (OUTHT) {
        _Float16 hp[8];
#pragma unroll
        for (int j = 0; j < 8; ++j) hp[j] = (_Float16)vals[j];
        *(u32x4*)(outHT + (size_t)c * M + rbase) = *(const u32x4*)hp;
      }
    }
  }
}

// ------------------------------------------------------------ attention
// Flash-style; 8 waves/block; K chunk [key][dk] + V^T chunk [dk][key] staged
// by TDM (double-buffered); all fragments loaded before WMMAs issue.
__global__ __launch_bounds__(256) void attn_kernel(
    const _Float16* __restrict__ Q, const _Float16* __restrict__ Kd,
    const _Float16* __restrict__ VT, const unsigned* __restrict__ maskb,
    _Float16* __restrict__ ctx) {
  __shared__ u32x4 Ks4[2][32 * 9];   // 32 keys x 64 halfs (+16B TDM pad)
  __shared__ u32x4 Vs4[2][64 * 5];   // 64 dk  x 32 halfs (+16B TDM pad)
  __shared__ u32x4 Ps4[8 * 16 * 5];  // per wave 16 x 32 halfs (+pad)

  int tid = threadIdx.x, wid = tid >> 5, lane = tid & 31;
  int lm = lane & 15, grp = lane >> 4;
  const int bps = S_LEN / 128;
  int b = blockIdx.x / (N_HEADS * bps);
  int rm = blockIdx.x % (N_HEADS * bps);
  int h = rm / bps;
  int q0 = (rm % bps) * 128 + wid * 16;
  long qrow = (long)b * S_LEN + q0;
  bool leader = (wid == 0);

  unsigned koff[2] = { (unsigned)(size_t)(void*)&Ks4[0][0],
                       (unsigned)(size_t)(void*)&Ks4[1][0] };
  unsigned voff[2] = { (unsigned)(size_t)(void*)&Vs4[0][0],
                       (unsigned)(size_t)(void*)&Vs4[1][0] };
  unsigned long long Kbase = (unsigned long long)(Kd + (size_t)b * S_LEN * D_MODEL + h * D_HEAD);
  unsigned long long Vbase = (unsigned long long)(VT + (size_t)h * D_HEAD * ROWS + (size_t)b * S_LEN);

  // Q fragments (DK chunks 0..31, 32..63)
  Frag16 qf[2];
  {
    const u32x4* qp = (const u32x4*)(Q + (qrow + lm) * D_MODEL + h * D_HEAD);
#pragma unroll
    for (int kc2 = 0; kc2 < 2; ++kc2) {
      qf[kc2].q[0] = qp[kc2 * 4 + grp * 2];
      qf[kc2].q[1] = qp[kc2 * 4 + grp * 2 + 1];
    }
  }

  float mrow[8], lrow[8];
#pragma unroll
  for (int j = 0; j < 8; ++j) { mrow[j] = -3.0e38f; lrow[j] = 0.0f; }
  v8f acc[4] = {};

  if (leader) {
    tdm_load_2d(koff[0], Kbase, 64, 32, D_MODEL, 4);  // rows of 32 dw + pad
    tdm_load_2d(voff[0], Vbase, 32, 64, ROWS, 3);     // rows of 16 dw + pad
    __builtin_amdgcn_s_wait_tensorcnt(0);
  }
  __syncthreads();

  int cur = 0;
  for (int kc = 0; kc < S_LEN; kc += 32) {
    bool more = (kc + 32) < S_LEN;
    if (more && leader) {
      tdm_load_2d(koff[1 - cur], Kbase + (size_t)(kc + 32) * D_MODEL * 2, 64, 32, D_MODEL, 4);
      tdm_load_2d(voff[1 - cur], Vbase + (size_t)(kc + 32) * 2, 32, 64, ROWS, 3);
    }
    unsigned mw[8];
#pragma unroll
    for (int j = 0; j < 8; ++j)
      mw[j] = maskb[(size_t)(q0 + j + grp * 8) * SW + (kc >> 5)];

    // all score B-fragments up front (distinct regs -> overlapped ds loads)
    Frag16 kb[2][2];
#pragma unroll
    for (int t = 0; t < 2; ++t) {
      const u32x4* kp4 = Ks4[cur] + (t * 16 + lm) * 9;
      kb[t][0].q[0] = kp4[grp * 2];     kb[t][0].q[1] = kp4[grp * 2 + 1];
      kb[t][1].q[0] = kp4[4 + grp * 2]; kb[t][1].q[1] = kp4[4 + grp * 2 + 1];
    }
    // V fragments early too (independent of softmax)
    Frag16 vfr[4];
#pragma unroll
    for (int t2 = 0; t2 < 4; ++t2) {
      const u32x4* vp4 = Vs4[cur] + (t2 * 16 + lm) * 5;
      vfr[t2].q[0] = vp4[grp * 2];
      vfr[t2].q[1] = vp4[grp * 2 + 1];
    }

    v8f sv[2];
#pragma unroll
    for (int t = 0; t < 2; ++t) {
      v8f s0 = {};
      s0 = __builtin_amdgcn_wmma_f32_16x16x32_f16(false, qf[0].v, false, kb[t][0].v, (short)0, s0, false, false);
      s0 = __builtin_amdgcn_wmma_f32_16x16x32_f16(false, qf[1].v, false, kb[t][1].v, (short)0, s0, false, false);
      sv[t] = s0;
    }

    float s[2][8];
#pragma unroll
    for (int t = 0; t < 2; ++t)
#pragma unroll
      for (int j = 0; j < 8; ++j) {
        float xv = sv[t][j] * 0.125f;                    // 1/sqrt(64)
        if (((mw[j] >> (t * 16 + lm)) & 1u) == 0u) xv = -1.0e9f;
        s[t][j] = xv;
      }

    _Float16* PsH = (_Float16*)(Ps4 + wid * 80);
#pragma unroll
    for (int j = 0; j < 8; ++j) {
      float mx = grp16_max(fmaxf(s[0][j], s[1][j]));
      float mnew = fmaxf(mrow[j], mx);
      float alpha = __expf(mrow[j] - mnew);
      float p0 = __expf(s[0][j] - mnew);
      float p1 = __expf(s[1][j] - mnew);
      lrow[j] = lrow[j] * alpha + grp16_sum(p0 + p1);
      mrow[j] = mnew;
#pragma unroll
      for (int t2 = 0; t2 < 4; ++t2) acc[t2][j] *= alpha;
      int prow = j + grp * 8;
      PsH[prow * 40 + lm]      = (_Float16)p0;
      PsH[prow * 40 + 16 + lm] = (_Float16)p1;
    }

    Frag16 pf;
    pf.q[0] = Ps4[wid * 80 + lm * 5 + grp * 2];
    pf.q[1] = Ps4[wid * 80 + lm * 5 + grp * 2 + 1];
#pragma unroll
    for (int t2 = 0; t2 < 4; ++t2)
      acc[t2] = __builtin_amdgcn_wmma_f32_16x16x32_f16(
          false, pf.v, false, vfr[t2].v, (short)0, acc[t2], false, false);

    if (more) {
      if (leader) __builtin_amdgcn_s_wait_tensorcnt(0);
      __syncthreads();
      cur ^= 1;
    }
  }

#pragma unroll
  for (int j = 0; j < 8; ++j) {
    float inv = 1.0f / lrow[j];
    size_t r = qrow + j + grp * 8;
#pragma unroll
    for (int t2 = 0; t2 < 4; ++t2)
      ctx[r * D_MODEL + h * D_HEAD + t2 * 16 + lm] = (_Float16)(acc[t2][j] * inv);
  }
}

// ------------------------------------------------------------ launch
extern "C" void kernel_launch(void* const* d_in, const int* in_sizes, int n_in,
                              void* d_out, int out_size, void* d_ws, size_t ws_size,
                              hipStream_t stream) {
  const float* x    = (const float*)d_in[0];
  const int*   mask = (const int*)d_in[1];
  const float* w_q  = (const float*)d_in[2];
  const float* w_k  = (const float*)d_in[3];
  const float* w_v  = (const float*)d_in[4];
  const float* w_o  = (const float*)d_in[5];
  const float* w1   = (const float*)d_in[6];
  const float* b1   = (const float*)d_in[7];
  const float* w2   = (const float*)d_in[8];
  const float* b2   = (const float*)d_in[9];
  const float* ln1a = (const float*)d_in[10];
  const float* ln1b = (const float*)d_in[11];
  const float* ln2a = (const float*)d_in[12];
  const float* ln2b = (const float*)d_in[13];
  float* out = (float*)d_out;

  char* p = (char*)d_ws;
  auto alloc = [&](size_t bytes) -> char* {
    char* r = p;
    p += (bytes + 255) & ~(size_t)255;
    return r;
  };
  const size_t DD = (size_t)D_MODEL * D_MODEL;
  const size_t DF = (size_t)D_MODEL * D_FF;
  const size_t RD = (size_t)ROWS * D_MODEL;
  const size_t RF = (size_t)ROWS * D_FF;

  _Float16* wqT = (_Float16*)alloc(DD * 2);
  _Float16* wkT = (_Float16*)alloc(DD * 2);
  _Float16* wvT = (_Float16*)alloc(DD * 2);
  _Float16* woT = (_Float16*)alloc(DD * 2);
  _Float16* w1T = (_Float16*)alloc(DF * 2);
  _Float16* w2T = (_Float16*)alloc(DF * 2);
  unsigned* mbit = (unsigned*)alloc((size_t)S_LEN * SW * 4);
  _Float16* h1_h = (_Float16*)alloc(RD * 2);
  _Float16* q_h  = (_Float16*)alloc(RD * 2);
  _Float16* k_h  = (_Float16*)alloc(RD * 2);
  _Float16* vT_h = (_Float16*)alloc(RD * 2);
  _Float16* cx_h = (_Float16*)alloc(RD * 2);
  float*    x1_f = (float*)   alloc(RD * 4);
  _Float16* h2_h = (_Float16*)alloc(RD * 2);
  _Float16* a1_h = (_Float16*)alloc(RF * 2);

  convT_kernel<<<(DD + 255) / 256, 256, 0, stream>>>(w_q, wqT, D_MODEL, D_MODEL);
  convT_kernel<<<(DD + 255) / 256, 256, 0, stream>>>(w_k, wkT, D_MODEL, D_MODEL);
  convT_kernel<<<(DD + 255) / 256, 256, 0, stream>>>(w_v, wvT, D_MODEL, D_MODEL);
  convT_kernel<<<(DD + 255) / 256, 256, 0, stream>>>(w_o, woT, D_MODEL, D_MODEL);
  convT_kernel<<<(DF + 255) / 256, 256, 0, stream>>>(w1, w1T, D_MODEL, D_FF);
  convT_kernel<<<(DF + 255) / 256, 256, 0, stream>>>(w2, w2T, D_FF, D_MODEL);
  mask_pack_kernel<<<(S_LEN * SW) / 8, 256, 0, stream>>>(mask, mbit);

  ln_kernel<<<ROWS / 8, 256, 0, stream>>>(x, ln1a, ln1b, h1_h);

  dim3 gD(D_MODEL / 64, ROWS / 128);
  dim3 gF(D_FF / 64, ROWS / 128);
  gemm_wmma_kernel<false, false, false, false, true, false><<<gD, 256, 0, stream>>>(
      h1_h, wqT, nullptr, nullptr, nullptr, q_h, nullptr, ROWS, D_MODEL, D_MODEL);
  gemm_wmma_kernel<false, false, false, false, true, false><<<gD, 256, 0, stream>>>(
      h1_h, wkT, nullptr, nullptr, nullptr, k_h, nullptr, ROWS, D_MODEL, D_MODEL);
  gemm_wmma_kernel<false, false, false, false, false, true><<<gD, 256, 0, stream>>>(
      h1_h, wvT, nullptr, nullptr, nullptr, nullptr, vT_h, ROWS, D_MODEL, D_MODEL);

  attn_kernel<<<BATCH * N_HEADS * (S_LEN / 128), 256, 0, stream>>>(q_h, k_h, vT_h, mbit, cx_h);

  gemm_wmma_kernel<false, false, true, true, false, false><<<gD, 256, 0, stream>>>(
      cx_h, woT, nullptr, x, x1_f, nullptr, nullptr, ROWS, D_MODEL, D_MODEL);

  ln_kernel<<<ROWS / 8, 256, 0, stream>>>(x1_f, ln2a, ln2b, h2_h);

  gemm_wmma_kernel<true, true, false, false, true, false><<<gF, 256, 0, stream>>>(
      h2_h, w1T, b1, nullptr, nullptr, a1_h, nullptr, ROWS, D_FF, D_MODEL);

  gemm_wmma_kernel<true, false, true, true, false, false><<<gD, 256, 0, stream>>>(
      a1_h, w2T, b2, x1_f, out, nullptr, nullptr, ROWS, D_MODEL, D_FF);
}
